// PatchFlow_10471130268090
// MI455X (gfx1250) — matrix operand order repaired and checked by reading, not compiled, not run on hardware
//
#include <hip/hip_runtime.h>

typedef __attribute__((ext_vector_type(16))) _Float16 v16h;
typedef __attribute__((ext_vector_type(8)))  _Float16 v8h;
typedef __attribute__((ext_vector_type(8)))  float    v8f;

#define N_ROWS   36864      // 96*96*4
#define N_PIX    9216       // 96*96
#define ISH      0.08838834764831845f  // 1/sqrt(128)

// ---------------- f16 weight blob sub-offsets (elements) ----------------
constexpr size_t O_SPWINID  = 0;                          // 4*128*32
constexpr size_t O_SPWINCTX = O_SPWINID  + 4*128*32;      // 4*128*128
constexpr size_t O_SPW1     = O_SPWINCTX + 4*128*128;     // 8*128*128
constexpr size_t O_SPW2     = O_SPW1     + 8*128*128;
constexpr size_t O_SPCW     = O_SPW2     + 8*128*128;
constexpr size_t O_SPWOUT   = O_SPCW     + 8*128*128;     // 4*192*128
constexpr size_t O_MW0      = O_SPWOUT   + 4*192*128;     // 128*32
constexpr size_t O_MCTXW    = O_MW0      + 128*32;        // 128*128
constexpr size_t O_MW1      = O_MCTXW    + 128*128;       // 2*128*128
constexpr size_t O_MW2      = O_MW1      + 2*128*128;
constexpr size_t O_MCW      = O_MW2      + 2*128*128;
constexpr size_t O_MWOUT    = O_MCW      + 2*128*128;     // 32*128
constexpr size_t O_EW1      = O_MWOUT    + 32*128;        // 128*128
constexpr size_t O_EW2      = O_EW1      + 128*128;       // 32*128
constexpr size_t W16_TOTAL  = O_EW2      + 32*128;

// ---------------- workspace byte offsets ----------------
constexpr size_t WS_CTX    = 0;                                   // 9216*128 f32
constexpr size_t WS_ZB     = WS_CTX    + (size_t)N_PIX*128*4;     // 36864*16 f32
constexpr size_t WS_MP     = WS_ZB     + (size_t)N_ROWS*16*4;     // 4*256 f32
constexpr size_t WS_LOGDET = WS_MP     + 4096;                    // 4 f32 (pad 256)
constexpr size_t WS_IDK    = WS_LOGDET + 256;                     // 4*8*43 f32 (pad)
constexpr size_t WS_W16    = WS_IDK    + 5632;                    // W16_TOTAL f16

// ---------------- device helpers ----------------
__device__ inline float softplusf(float x) { return x > 20.f ? x : log1pf(__expf(x)); }

__device__ inline void mk_knots(const float* u, float* cv, float* w) {
  float mx = u[0];
  #pragma unroll
  for (int k = 1; k < 8; ++k) mx = fmaxf(mx, u[k]);
  float e[8], s = 0.f;
  #pragma unroll
  for (int k = 0; k < 8; ++k) { e[k] = __expf(u[k] - mx); s += e[k]; }
  float inv = 1.f / s;
  cv[0] = -3.0f;
  float cum = 0.f;
  #pragma unroll
  for (int k = 0; k < 8; ++k) {
    float v = 0.001f + (1.f - 0.008f) * e[k] * inv;
    cum += v;
    cv[k + 1] = cum * 6.f - 3.f;
  }
  cv[8] = 3.0f;
  #pragma unroll
  for (int k = 0; k < 8; ++k) w[k] = cv[k + 1] - cv[k];
}

__device__ inline void rqeval(float x, const float* cw, const float* wd,
                              const float* ch, const float* hh, const float* dd,
                              float& yo, float& ldo) {
  bool inside = (x >= -3.0f) && (x <= 3.0f);
  float xc = fminf(fmaxf(x, -3.0f), 3.0f);
  int cnt = 0;
  #pragma unroll
  for (int k = 0; k < 9; ++k) cnt += (xc >= cw[k]) ? 1 : 0;
  int i = cnt - 1; i = i < 0 ? 0 : (i > 7 ? 7 : i);
  float cwk = cw[0], wdk = wd[0], chk = ch[0], hhk = hh[0], dk = dd[0], dk1 = dd[1];
  #pragma unroll
  for (int k = 1; k < 8; ++k) {
    bool c = (i == k);
    cwk = c ? cw[k] : cwk; wdk = c ? wd[k]   : wdk;
    chk = c ? ch[k] : chk; hhk = c ? hh[k]   : hhk;
    dk  = c ? dd[k] : dk;  dk1 = c ? dd[k+1] : dk1;
  }
  float th  = (xc - cwk) / wdk;
  float t1  = th * (1.f - th);
  float dl  = hhk / wdk;
  float den = dl + (dk + dk1 - 2.f * dl) * t1;
  float y   = chk + hhk * (dl * th * th + dk * t1) / den;
  float omt = 1.f - th;
  float dnum = dl * dl * (dk1 * th * th + 2.f * dl * t1 + dk * omt * omt);
  float ld = logf(dnum) - 2.f * logf(den);
  yo  = inside ? y  : x;
  ldo = inside ? ld : 0.f;
}

// --- WMMA fragment loaders (CDNA5 16x16x32 f16 layouts, ISA 7.12.2) ---
__device__ inline v16h packAB(const _Float16* lo, const _Float16* hi) {
  v8h a = *(const v8h*)lo;
  v8h b = *(const v8h*)hi;
  v16h r;
  #pragma unroll
  for (int j = 0; j < 8; ++j) { r[j] = a[j]; r[j + 8] = b[j]; }
  return r;
}
// A: lanes 0-15 hold M=0-15 (K 0..7,16..23); lanes 16-31 same rows (K 8..15,24..31)
__device__ inline v16h loadA(const _Float16* x, int xs, int k0, int lane) {
  int m = lane & 15, kh = (lane >> 4) << 3;
  const _Float16* p = x + m * xs + k0 + kh;
  return packAB(p, p + 16);
}
// B: lanes 0-15 hold K 0..7,16..23; lanes 16-31 K 8..15,24..31, per-lane col n = lane&15 (+16*nt)
__device__ inline v16h loadB(const _Float16* W, int ks, int nt, int k0, int lane) {
  int n = (lane & 15) + (nt << 4), kh = (lane >> 4) << 3;
  const _Float16* p = W + (size_t)n * ks + k0 + kh;
  return packAB(p, p + 16);
}

template<int NT, int KS>
__device__ inline void gemm(v8f* acc, const _Float16* x, int xs,
                            const _Float16* W, int wks, int lane) {
  for (int k = 0; k < KS; ++k) {
    v16h a = loadA(x, xs, k * 32, lane);
    #pragma unroll
    for (int nt = 0; nt < NT; ++nt) {
      v16h b = loadB(W, wks, nt, k * 32, lane);
      acc[nt] = __builtin_amdgcn_wmma_f32_16x16x32_f16(
          false, a, false, b, (short)0, acc[nt], false, false);
    }
  }
}

template<int NT>
__device__ inline void storeF16(const v8f* acc, _Float16* dst, int stride, int lane, bool relu) {
  int ms = (lane >> 4) << 3, n0 = lane & 15;
  #pragma unroll
  for (int nt = 0; nt < NT; ++nt)
    #pragma unroll
    for (int r = 0; r < 8; ++r) {
      float v = acc[nt][r];
      if (relu) v = fmaxf(v, 0.f);
      dst[(r + ms) * stride + n0 + (nt << 4)] = (_Float16)v;
    }
}
template<int NT>
__device__ inline void storeF32(const v8f* acc, float* dst, int stride, int coff, int lane) {
  int ms = (lane >> 4) << 3, n0 = lane & 15;
  #pragma unroll
  for (int nt = 0; nt < NT; ++nt)
    #pragma unroll
    for (int r = 0; r < 8; ++r)
      dst[(r + ms) * stride + coff + n0 + (nt << 4)] = acc[nt][r];
}
template<int NT>
__device__ inline void initBias(v8f* acc, const float* b, int lane) {
  int n0 = lane & 15;
  #pragma unroll
  for (int nt = 0; nt < NT; ++nt) {
    float bv = b[n0 + (nt << 4)];
    #pragma unroll
    for (int r = 0; r < 8; ++r) acc[nt][r] = bv;
  }
}

// u = relu(t)@W1^T+b1 ; u = relu(u)@W2^T+b2 ; t += u * sigmoid(ctx@CW^T + Cb)
__device__ inline void resblock(v8f* tacc, _Float16* act, const _Float16* ctxh,
                                const _Float16* W1, const _Float16* W2, const _Float16* CW,
                                const float* b1, const float* b2, const float* cb, int lane) {
  storeF16<8>(tacc, act, 128, lane, true);
  v8f u[8];  initBias<8>(u, b1, lane);
  gemm<8, 4>(u, act, 128, W1, 128, lane);
  storeF16<8>(u, act, 128, lane, true);
  v8f u2[8]; initBias<8>(u2, b2, lane);
  gemm<8, 4>(u2, act, 128, W2, 128, lane);
  v8f g[8];  initBias<8>(g, cb, lane);
  gemm<8, 4>(g, ctxh, 128, CW, 128, lane);
  #pragma unroll
  for (int nt = 0; nt < 8; ++nt)
    #pragma unroll
    for (int r = 0; r < 8; ++r)
      tacc[nt][r] += u2[nt][r] * (1.f / (1.f + __expf(-g[nt][r])));
}

// ======================= prep kernels =======================
__global__ void k_spatial(const float* __restrict__ x, float* __restrict__ zb) {
  int t = blockIdx.x * blockDim.x + threadIdx.x;
  if (t >= N_ROWS * 16) return;
  int s = t & 15, r = t >> 4;
  int b = r & 3, pix = r >> 2;
  int oh = pix / 96, ow = pix % 96;
  const float* xb = x + ((size_t)(b * 16 + s)) * 3 * 192 * 192;
  float acc = 0.f;
  for (int kc = 0; kc < 3; ++kc)
    for (int kh = 0; kh < 3; ++kh) {
      int ih = oh * 2 + kh - 1;
      if (ih < 0 || ih >= 192) continue;
      for (int kw = 0; kw < 3; ++kw) {
        int iw = ow * 2 + kw - 1;
        if (iw < 0 || iw >= 192) continue;
        float v = xb[(kc * 192 + ih) * 192 + iw];
        acc += v * v;
      }
    }
  zb[r * 16 + s] = sqrtf(acc);
}

__global__ void k_posenc(float* __restrict__ ctx) {
  int t = blockIdx.x * blockDim.x + threadIdx.x;
  if (t >= N_PIX * 128) return;
  int c = t & 127, pix = t >> 7;
  int oh = pix / 96, ow = pix % 96;
  int cc = c & 63;
  int pos = (c < 64) ? oh : ow;
  int j = cc >> 1;
  float f = __expf(-(float)(2 * j) / 64.f * 9.210340371976184f); // ln(1e4)
  float ph = (float)pos * f;
  ctx[t] = (cc & 1) ? __cosf(ph) : __sinf(ph);
}

__global__ void k_prep_lu(const float* __restrict__ lo, const float* __restrict__ up,
                          const float* __restrict__ ud, const int* __restrict__ perm,
                          float* __restrict__ Mp, float* __restrict__ logdet) {
  int t = blockIdx.x * blockDim.x + threadIdx.x;
  if (t < 4 * 256) {
    int l = t >> 8, o = (t >> 4) & 15, k = t & 15;
    float acc = 0.f;
    int jmax = o < k ? o : k;
    for (int j = 0; j <= jmax; ++j) {
      float Lv = (j < o) ? lo[(l * 16 + o) * 16 + j] : 1.f;
      float Uv = (j < k) ? up[(l * 16 + j) * 16 + k]
                         : (softplusf(ud[l * 16 + k]) + 0.001f);
      acc += Lv * Uv;
    }
    Mp[l * 256 + o * 16 + perm[l * 16 + k]] = acc;
  }
  if (t < 4) {
    float s = 0.f;
    for (int k = 0; k < 16; ++k) s += logf(softplusf(ud[t * 16 + k]) + 0.001f);
    logdet[t] = s;
  }
}

__global__ void k_prep_knots(const float* __restrict__ uw, const float* __restrict__ uh,
                             const float* __restrict__ udp, float* __restrict__ idk) {
  int t = blockIdx.x * blockDim.x + threadIdx.x;
  if (t >= 32) return;
  int l = t >> 3, di = t & 7;
  float u[8], cv[9], w[8];
  float* kb = idk + (l * 8 + di) * 43;
  for (int j = 0; j < 8; ++j) u[j] = uw[(l * 8 + di) * 8 + j];
  mk_knots(u, cv, w);
  for (int j = 0; j < 9; ++j) kb[j] = cv[j];
  for (int j = 0; j < 8; ++j) kb[9 + j] = w[j];
  for (int j = 0; j < 8; ++j) u[j] = uh[(l * 8 + di) * 8 + j];
  mk_knots(u, cv, w);
  for (int j = 0; j < 9; ++j) kb[17 + j] = cv[j];
  for (int j = 0; j < 8; ++j) kb[26 + j] = w[j];
  kb[34] = 1.f; kb[42] = 1.f;
  for (int j = 1; j < 8; ++j) kb[34 + j] = 0.001f + softplusf(udp[(l * 8 + di) * 7 + (j - 1)]);
}

__global__ void k_prep_w16(const float* spWin, const float* spW1, const float* spW2,
                           const float* spCW, const float* spWout,
                           const float* mW0, const float* mCtxW, const float* mW1,
                           const float* mW2, const float* mCW, const float* mWout,
                           const float* eW1, const float* eW2, _Float16* w) {
  int tid = blockIdx.x * blockDim.x + threadIdx.x;
  int stride = gridDim.x * blockDim.x;
  for (int i = tid; i < 4 * 128 * 32; i += stride) {
    int l = i >> 12, r = (i >> 5) & 127, c = i & 31;
    w[O_SPWINID + i] = (_Float16)(c < 8 ? spWin[(l * 128 + r) * 136 + c] : 0.f);
  }
  for (int i = tid; i < 4 * 128 * 128; i += stride) {
    int l = i >> 14, r = (i >> 7) & 127, c = i & 127;
    w[O_SPWINCTX + i] = (_Float16)spWin[(l * 128 + r) * 136 + 8 + c];
  }
  for (int i = tid; i < 8 * 128 * 128; i += stride) w[O_SPW1 + i] = (_Float16)spW1[i];
  for (int i = tid; i < 8 * 128 * 128; i += stride) w[O_SPW2 + i] = (_Float16)spW2[i];
  for (int i = tid; i < 8 * 128 * 128; i += stride) w[O_SPCW + i] = (_Float16)spCW[i];
  for (int i = tid; i < 4 * 192 * 128; i += stride) {
    int l = i / (192 * 128), rem = i - l * 192 * 128, r = rem >> 7, c = rem & 127;
    w[O_SPWOUT + i] = (_Float16)(r < 184 ? spWout[(l * 184 + r) * 128 + c] : 0.f);
  }
  for (int i = tid; i < 128 * 32; i += stride) {
    int r = i >> 5, c = i & 31;
    float v = 0.f;
    if (c < 16 && ((r % 15) + 1) >= (c + 1)) v = mW0[r * 16 + c];   // MASK_IN
    w[O_MW0 + i] = (_Float16)v;
  }
  for (int i = tid; i < 128 * 128; i += stride) w[O_MCTXW + i] = (_Float16)mCtxW[i];
  for (int i = tid; i < 2 * 128 * 128; i += stride) {
    int r = (i >> 7) & 127, c = i & 127;
    float m = (((r % 15) + 1) >= ((c % 15) + 1)) ? 1.f : 0.f;       // MASK_HID
    w[O_MW1 + i] = (_Float16)(mW1[i] * m);
    w[O_MW2 + i] = (_Float16)(mW2[i] * m);
  }
  for (int i = tid; i < 2 * 128 * 128; i += stride) w[O_MCW + i] = (_Float16)mCW[i];
  for (int i = tid; i < 32 * 128; i += stride) {
    int o = i >> 7, j = i & 127;
    float m = (((o % 16) + 1) > ((j % 15) + 1)) ? 1.f : 0.f;        // MASK_OUT
    w[O_MWOUT + i] = (_Float16)(mWout[i] * m);
  }
  for (int i = tid; i < 128 * 128; i += stride) w[O_EW1 + i] = (_Float16)eW1[i];
  for (int i = tid; i < 32 * 128; i += stride) w[O_EW2 + i] = (_Float16)eW2[i];
}

// ======================= fused flow kernel =======================
__global__ __launch_bounds__(64)
void flow_main(const float* __restrict__ ctx, const float* __restrict__ zb,
               float* __restrict__ out,
               const float* __restrict__ Mp, const float* __restrict__ logdet,
               const float* __restrict__ idk, const _Float16* __restrict__ w16,
               const float* lu_bias, const float* sp_bin, const float* sp_bb1,
               const float* sp_bb2, const float* sp_bCb, const float* sp_bout,
               const float* made_b0, const float* made_ctxb, const float* made_bb1,
               const float* made_bb2, const float* made_bCb, const float* made_bout,
               const float* enc_b1, const float* enc_b2) {
  __shared__ __attribute__((aligned(16))) _Float16 s_ctxh[2][16 * 128];
  __shared__ __attribute__((aligned(16))) _Float16 s_act[2][16 * 128];
  __shared__ __attribute__((aligned(16))) _Float16 s_zin[2][16 * 32];
  __shared__ float s_z[2][2][16 * 16];
  __shared__ __attribute__((aligned(16))) float s_pbf[2][16 * 96];
  __shared__ float s_lq[2][16];

  const int lane = threadIdx.x & 31, wave = threadIdx.x >> 5;
  const int tile = blockIdx.x * 2 + wave;
  const int row0 = tile * 16;

  _Float16* ctxh = &s_ctxh[wave][0];
  _Float16* act  = &s_act[wave][0];
  _Float16* zin  = &s_zin[wave][0];
  _Float16* pbh  = reinterpret_cast<_Float16*>(&s_pbf[wave][0]);  // 16x192 f16 view
  float* zcur = &s_z[wave][0][0];
  float* znxt = &s_z[wave][1][0];

  // init: ctx (f16), z, per-row logq
  for (int idx = lane; idx < 16 * 128; idx += 32) {
    int m = idx >> 7, c = idx & 127;
    int pix = (row0 + m) >> 2;
    ctxh[idx] = (_Float16)ctx[(size_t)pix * 128 + c];
  }
  for (int idx = lane; idx < 256; idx += 32) zcur[idx] = zb[(size_t)row0 * 16 + idx];
  if (lane < 16) s_lq[wave][lane] = 0.f;
  __syncthreads();

  // ---------------- 4 coupling layers ----------------
  for (int li = 3; li >= 0; --li) {
    // permute + LU:  z' = z @ Mp^T + bias
    const float* mp = Mp + li * 256;
    const float* lb = lu_bias + li * 16;
    for (int idx = lane; idx < 256; idx += 32) {
      int m = idx >> 4, o = idx & 15;
      float s = lb[o];
      #pragma unroll
      for (int k = 0; k < 16; ++k) s += mp[o * 16 + k] * zcur[m * 16 + k];
      znxt[m * 16 + o] = s;
    }
    { float* t = zcur; zcur = znxt; znxt = t; }
    if (lane < 16) s_lq[wave][lane] += logdet[li];
    __syncthreads();

    // zid -> f16 A tile (K padded to 32)
    for (int idx = lane; idx < 512; idx += 32) zin[idx] = (_Float16)0.f;
    for (int idx = lane; idx < 128; idx += 32) {
      int m = idx >> 3, d = idx & 7;
      zin[m * 32 + d] = (_Float16)zcur[m * 16 + 2 * d];
    }

    // t = [zid|ctx] @ Win^T + bin
    v8f tacc[8]; initBias<8>(tacc, sp_bin + li * 128, lane);
    gemm<8, 1>(tacc, zin, 32, w16 + O_SPWINID + (size_t)li * 128 * 32, 32, lane);
    gemm<8, 4>(tacc, ctxh, 128, w16 + O_SPWINCTX + (size_t)li * 128 * 128, 128, lane);
    for (int blk = 0; blk < 2; ++blk) {
      int bi = li * 2 + blk;
      resblock(tacc, act, ctxh,
               w16 + O_SPW1 + (size_t)bi * 16384,
               w16 + O_SPW2 + (size_t)bi * 16384,
               w16 + O_SPCW + (size_t)bi * 16384,
               sp_bb1 + bi * 128, sp_bb2 + bi * 128, sp_bCb + bi * 128, lane);
    }
    storeF16<8>(tacc, act, 128, lane, false);

    // p = t @ Wout^T + bout   (N padded 184 -> 192)
    v8f pacc[12];
    {
      int n0 = lane & 15;
      #pragma unroll
      for (int nt = 0; nt < 12; ++nt) {
        int n = n0 + (nt << 4);
        float bv = (n < 184) ? sp_bout[li * 184 + n] : 0.f;
        #pragma unroll
        for (int r = 0; r < 8; ++r) pacc[nt][r] = bv;
      }
    }
    gemm<12, 4>(pacc, act, 128, w16 + O_SPWOUT + (size_t)li * 192 * 128, 128, lane);
    __syncthreads();
    storeF16<12>(pacc, pbh, 192, lane, false);
    __syncthreads();

    // splines: transform dims (odd) use p; identity dims (even) use precomputed knots
    for (int idx = lane; idx < 128; idx += 32) {
      int m = idx >> 3, di = idx & 7;
      const _Float16* pr = pbh + m * 192 + di * 23;
      float uw[8], uh[8], dd[9];
      #pragma unroll
      for (int j = 0; j < 8; ++j) { uw[j] = ISH * (float)pr[j]; uh[j] = ISH * (float)pr[8 + j]; }
      dd[0] = 1.f; dd[8] = 1.f;
      #pragma unroll
      for (int j = 1; j < 8; ++j) dd[j] = 0.001f + softplusf((float)pr[16 + j - 1]);
      float cw[9], wd[8], ch[9], hh[8];
      mk_knots(uw, cw, wd);
      mk_knots(uh, ch, hh);
      float y, ld;
      rqeval(zcur[m * 16 + 2 * di + 1], cw, wd, ch, hh, dd, y, ld);
      znxt[m * 16 + 2 * di + 1] = y;

      const float* kb = idk + (li * 8 + di) * 43;
      float cw2[9], wd2[8], ch2[9], hh2[8], dd2[9];
      #pragma unroll
      for (int j = 0; j < 9; ++j) { cw2[j] = kb[j]; ch2[j] = kb[17 + j]; dd2[j] = kb[34 + j]; }
      #pragma unroll
      for (int j = 0; j < 8; ++j) { wd2[j] = kb[9 + j]; hh2[j] = kb[26 + j]; }
      float y2, ld2;
      rqeval(zcur[m * 16 + 2 * di], cw2, wd2, ch2, hh2, dd2, y2, ld2);
      znxt[m * 16 + 2 * di] = y2;
      atomicAdd(&s_lq[wave][m], ld + ld2);
    }
    { float* t = zcur; zcur = znxt; znxt = t; }
    __syncthreads();
  }

  // ---------------- MADE ----------------
  for (int idx = lane; idx < 512; idx += 32) zin[idx] = (_Float16)0.f;
  for (int idx = lane; idx < 256; idx += 32) {
    int m = idx >> 4, d = idx & 15;
    zin[m * 32 + d] = (_Float16)zcur[m * 16 + d];
  }
  v8f tacc[8];
  {
    int n0 = lane & 15;
    #pragma unroll
    for (int nt = 0; nt < 8; ++nt) {
      int n = n0 + (nt << 4);
      float bv = made_b0[n] + made_ctxb[n];
      #pragma unroll
      for (int r = 0; r < 8; ++r) tacc[nt][r] = bv;
    }
  }
  gemm<8, 1>(tacc, zin, 32, w16 + O_MW0, 32, lane);
  gemm<8, 4>(tacc, ctxh, 128, w16 + O_MCTXW, 128, lane);
  for (int blk = 0; blk < 2; ++blk) {
    resblock(tacc, act, ctxh,
             w16 + O_MW1 + (size_t)blk * 16384,
             w16 + O_MW2 + (size_t)blk * 16384,
             w16 + O_MCW + (size_t)blk * 16384,
             made_bb1 + blk * 128, made_bb2 + blk * 128, made_bCb + blk * 128, lane);
  }
  storeF16<8>(tacc, act, 128, lane, false);
  v8f ar[2]; initBias<2>(ar, made_bout, lane);
  gemm<2, 4>(ar, act, 128, w16 + O_MWOUT, 128, lane);
  __syncthreads();
  storeF32<2>(ar, &s_pbf[wave][0], 96, 0, lane);
  __syncthreads();

  // affine head: z = softplus(ar0)+eps * z + ar1
  for (int idx = lane; idx < 256; idx += 32) {
    int m = idx >> 4, d = idx & 15;
    float sc = softplusf(s_pbf[wave][m * 96 + 2 * d]) + 0.001f;
    float sh = s_pbf[wave][m * 96 + 2 * d + 1];
    znxt[m * 16 + d] = sc * zcur[m * 16 + d] + sh;
    atomicAdd(&s_lq[wave][m], logf(sc));
  }
  { float* t = zcur; zcur = znxt; znxt = t; }
  __syncthreads();

  // encoder: e = silu(ctx@W1^T+b1) @ W2^T + b2
  v8f hacc[8]; initBias<8>(hacc, enc_b1, lane);
  gemm<8, 4>(hacc, ctxh, 128, w16 + O_EW1, 128, lane);
  #pragma unroll
  for (int nt = 0; nt < 8; ++nt)
    #pragma unroll
    for (int r = 0; r < 8; ++r) {
      float v = hacc[nt][r];
      hacc[nt][r] = v / (1.f + __expf(-v));
    }
  storeF16<8>(hacc, act, 128, lane, false);
  v8f eacc[2]; initBias<2>(eacc, enc_b2, lane);
  gemm<2, 4>(eacc, act, 128, w16 + O_EW2, 128, lane);
  __syncthreads();
  storeF32<2>(eacc, &s_pbf[wave][0], 96, 32, lane);
  __syncthreads();

  // Gaussian log-prob + writeout
  if (lane < 16) {
    int m = lane;
    float acc = s_lq[wave][m] - 14.70301652529297f;  // -0.5*16*ln(2*pi)
    const float* pm = &s_pbf[wave][m * 96 + 32];
    #pragma unroll
    for (int d = 0; d < 16; ++d) {
      float zv = zcur[m * 16 + d];
      float mean = pm[d], ls = pm[16 + d];
      float u = (zv - mean) * __expf(-ls);
      acc -= ls + 0.5f * u * u;
    }
    int row = row0 + m;
    out[(size_t)(row & 3) * N_PIX + (row >> 2)] = acc;
  }
}

// ======================= launcher =======================
extern "C" void kernel_launch(void* const* d_in, const int* in_sizes, int n_in,
                              void* d_out, int out_size, void* d_ws, size_t ws_size,
                              hipStream_t stream) {
  (void)in_sizes; (void)n_in; (void)out_size; (void)ws_size;
  char* ws = (char*)d_ws;
  float*    ctx    = (float*)(ws + WS_CTX);
  float*    zbuf   = (float*)(ws + WS_ZB);
  float*    Mp     = (float*)(ws + WS_MP);
  float*    logdet = (float*)(ws + WS_LOGDET);
  float*    idk    = (float*)(ws + WS_IDK);
  _Float16* w16    = (_Float16*)(ws + WS_W16);

  const float* x        = (const float*)d_in[0];
  const float* made_W0  = (const float*)d_in[1];
  const float* made_b0  = (const float*)d_in[2];
  const float* made_ctxW= (const float*)d_in[3];
  const float* made_ctxb= (const float*)d_in[4];
  const float* made_bW1 = (const float*)d_in[5];
  const float* made_bb1 = (const float*)d_in[6];
  const float* made_bW2 = (const float*)d_in[7];
  const float* made_bb2 = (const float*)d_in[8];
  const float* made_bCW = (const float*)d_in[9];
  const float* made_bCb = (const float*)d_in[10];
  const float* made_Wout= (const float*)d_in[11];
  const float* made_bout= (const float*)d_in[12];
  const float* sp_Win   = (const float*)d_in[13];
  const float* sp_bin   = (const float*)d_in[14];
  const float* sp_bW1   = (const float*)d_in[15];
  const float* sp_bb1   = (const float*)d_in[16];
  const float* sp_bW2   = (const float*)d_in[17];
  const float* sp_bb2   = (const float*)d_in[18];
  const float* sp_bCW   = (const float*)d_in[19];
  const float* sp_bCb   = (const float*)d_in[20];
  const float* sp_Wout  = (const float*)d_in[21];
  const float* sp_bout  = (const float*)d_in[22];
  const float* sp_uw    = (const float*)d_in[23];
  const float* sp_uh    = (const float*)d_in[24];
  const float* sp_ud    = (const float*)d_in[25];
  const float* lu_lower = (const float*)d_in[26];
  const float* lu_upper = (const float*)d_in[27];
  const float* lu_udiag = (const float*)d_in[28];
  const float* lu_bias  = (const float*)d_in[29];
  const int*   lu_perm  = (const int*)d_in[30];
  const float* enc_W1   = (const float*)d_in[31];
  const float* enc_b1   = (const float*)d_in[32];
  const float* enc_W2   = (const float*)d_in[33];
  const float* enc_b2   = (const float*)d_in[34];
  float* out = (float*)d_out;

  k_prep_w16<<<256, 256, 0, stream>>>(sp_Win, sp_bW1, sp_bW2, sp_bCW, sp_Wout,
                                      made_W0, made_ctxW, made_bW1, made_bW2,
                                      made_bCW, made_Wout, enc_W1, enc_W2, w16);
  k_prep_lu<<<4, 256, 0, stream>>>(lu_lower, lu_upper, lu_udiag, lu_perm, Mp, logdet);
  k_prep_knots<<<1, 32, 0, stream>>>(sp_uw, sp_uh, sp_ud, idk);
  k_spatial<<<(N_ROWS * 16 + 255) / 256, 256, 0, stream>>>(x, zbuf);
  k_posenc<<<(N_PIX * 128 + 255) / 256, 256, 0, stream>>>(ctx);
  flow_main<<<N_ROWS / 32, 64, 0, stream>>>(
      ctx, zbuf, out, Mp, logdet, idk, w16,
      lu_bias, sp_bin, sp_bb1, sp_bb2, sp_bCb, sp_bout,
      made_b0, made_ctxb, made_bb1, made_bb2, made_bCb, made_bout,
      enc_b1, enc_b2);
}
